// KVMemoryLayer_49555332661705
// MI455X (gfx1250) — matrix operand order repaired and checked by reading, not compiled, not run on hardware
//
#include <hip/hip_runtime.h>
#include <math.h>

typedef __bf16 bf16_t;
typedef bf16_t v16bf __attribute__((ext_vector_type(16)));
typedef float  v8f   __attribute__((ext_vector_type(8)));

#define NQ      512
#define DIM     1024
#define NS      200000
#define KTOP    32
#define QG      2                  // query 16-tiles per block
#define QBLK    (QG * 16)          // 32 queries per block
#define NQG     (NQ / QBLK)        // 16 query groups
#define CHUNKS  32
#define CH      (NS / CHUNKS)      // 6250 slots per chunk (exact)
#define TILES   ((CH + 15) / 16)   // 391 16-slot tiles per chunk
#define NWAVE   8                  // 256 threads = 8 wave32

struct Cand { float v; int i; };

// ---------------------------------------------------------------------------
// Wave-cooperative: extract top-32 (descending) of 32*NPL candidates, NPL/lane.
// ---------------------------------------------------------------------------
template <int NPL>
__device__ __forceinline__ void wave_selectN(float (&cv)[NPL], int (&ci)[NPL],
                                             int lane, float* outV, int* outI) {
  for (int s = 0; s < KTOP; ++s) {
    float bm = cv[0]; int bi = 0;
#pragma unroll
    for (int i = 1; i < NPL; ++i) if (cv[i] > bm) { bm = cv[i]; bi = i; }
    float M = bm; int Ml = lane;
#pragma unroll
    for (int off = 16; off; off >>= 1) {
      float om = __shfl_xor(M, off, 32);
      int  oml = __shfl_xor(Ml, off, 32);
      if (om > M || (om == M && oml < Ml)) { M = om; Ml = oml; }
    }
    if (lane == Ml) {
      int chosen = ci[0];
#pragma unroll
      for (int i = 1; i < NPL; ++i) chosen = (bi == i) ? ci[i] : chosen;
      outV[s] = bm;
      outI[s] = chosen;
#pragma unroll
      for (int i = 0; i < NPL; ++i) if (bi == i) cv[i] = -__builtin_inff();
    }
  }
}

// ---------------------------------------------------------------------------
// LDS-resident top-32 insert. 32-entry list, entry #lane per lane. Divergent
// STORES (not register phis) carry the update -> no allocator copy storms.
// Returns the list's new minimum (broadcast in all lanes).
// ---------------------------------------------------------------------------
__device__ __forceinline__ float lds_insert(float* lv, int* li,
                                            float v, int slotIdx, int lane) {
  const float e = lv[lane];
  float m = e; int ml = lane;
#pragma unroll
  for (int off = 16; off; off >>= 1) {
    float om = __shfl_xor(m, off, 32);
    int  oml = __shfl_xor(ml, off, 32);
    const bool take = (om < m) || (om == m && oml < ml);
    m  = take ? om : m;
    ml = take ? oml : ml;
  }
  const bool accept = v > m;
  const bool repl   = accept && (lane == ml);
  if (repl) { lv[lane] = v; li[lane] = slotIdx; }
  float m2 = repl ? v : e;
#pragma unroll
  for (int off = 16; off; off >>= 1) m2 = fminf(m2, __shfl_xor(m2, off, 32));
  return accept ? m2 : m;
}

// ---------------------------------------------------------------------------
// Kernel 1: WMMA scores (bf16 in / f32 acc), 2 query tiles per B fragment,
// streamed LDS-resident top-32. grid = (NQG, CHUNKS), block = 256.
// LDS: 64 KB q-tile + 64 KB lists = 128 KB (<320 KB/WG per CDNA5 ISA).
// ---------------------------------------------------------------------------
__global__ __launch_bounds__(256, 1)
void score_topk_kernel(const float* __restrict__ x,
                       const float* __restrict__ keys,
                       Cand* __restrict__ cands) {
  __shared__ __align__(16) unsigned short qs[QBLK * DIM];     // 64 KB bf16 q-tile
  __shared__ float listV[NWAVE * QBLK * KTOP];                // 32 KB
  __shared__ int   listI[NWAVE * QBLK * KTOP];                // 32 KB

  const int tid  = threadIdx.x;
  const int lane = tid & 31;
  const int wv   = tid >> 5;
  const int qg   = blockIdx.x;
  const int chunk = blockIdx.y;
  const int chunkStart = chunk * CH;
  const int chunkEnd   = chunkStart + CH;

  // Load 32x1024 q block, convert f32 -> bf16 into LDS.
  const float* xq = x + (size_t)qg * QBLK * DIM;
  for (int n = tid; n < QBLK * DIM; n += 256) {
    union { bf16_t h; unsigned short u; } cvt;
    cvt.h = (bf16_t)xq[n];
    qs[n] = cvt.u;
  }
  for (int n = tid; n < NWAVE * QBLK * KTOP; n += 256) {
    listV[n] = -__builtin_inff();
    listI[n] = 0;
  }
  __syncthreads();

  float thr[QG][8];
#pragma unroll
  for (int tt = 0; tt < QG; ++tt)
#pragma unroll
    for (int r = 0; r < 8; ++r) thr[tt][r] = -__builtin_inff();

  const int row = lane & 15;       // B column (slot within tile) / A row
  const int hi  = lane >> 4;
  const int aLo = hi * 16;         // A K-offset (contiguous, matches B)
  const int bKo = hi * 16;         // B K-offset (contiguous, matches A)

  float* myV = listV + wv * QBLK * KTOP;
  int*   myI = listI + wv * QBLK * KTOP;

  for (int t = wv; t < TILES; t += NWAVE) {
    const int slotBase = chunkStart + t * 16;
    const int slot     = slotBase + row;
    const bool valid   = slot < chunkEnd;
    const int cslot    = valid ? slot : (chunkEnd - 1);
    const float* krow  = keys + (size_t)cslot * DIM;

    if (t + NWAVE < TILES) {
      __builtin_prefetch(keys + (size_t)(chunkStart + (t + NWAVE) * 16 + row) * DIM, 0, 1);
    }

    v8f cA[QG];
#pragma unroll
    for (int tt = 0; tt < QG; ++tt) cA[tt] = (v8f){};

    for (int ks = 0; ks < DIM / 32; ++ks) {
      const int kk = ks * 32;
      // B fragment: 16 contiguous f32 of this lane's key row -> bf16.
      union { v16bf v; bf16_t h[16]; } B;
      const float4* kp = (const float4*)(krow + kk + bKo);
      float4 f0 = kp[0], f1 = kp[1], f2 = kp[2], f3 = kp[3];
      B.h[0]  = (bf16_t)f0.x; B.h[1]  = (bf16_t)f0.y;
      B.h[2]  = (bf16_t)f0.z; B.h[3]  = (bf16_t)f0.w;
      B.h[4]  = (bf16_t)f1.x; B.h[5]  = (bf16_t)f1.y;
      B.h[6]  = (bf16_t)f1.z; B.h[7]  = (bf16_t)f1.w;
      B.h[8]  = (bf16_t)f2.x; B.h[9]  = (bf16_t)f2.y;
      B.h[10] = (bf16_t)f2.z; B.h[11] = (bf16_t)f2.w;
      B.h[12] = (bf16_t)f3.x; B.h[13] = (bf16_t)f3.y;
      B.h[14] = (bf16_t)f3.z; B.h[15] = (bf16_t)f3.w;
      // One B fragment feeds QG WMMAs (A from LDS, two b128 each).
#pragma unroll
      for (int tt = 0; tt < QG; ++tt) {
        union { v16bf v; uint4 q[2]; } A;
        A.q[0] = *(const uint4*)(qs + (tt * 16 + row) * DIM + kk + aLo);
        A.q[1] = *(const uint4*)(qs + (tt * 16 + row) * DIM + kk + aLo + 8);
        cA[tt] = __builtin_amdgcn_wmma_f32_16x16x32_bf16(false, A.v, false, B.v,
                                                         (short)0, cA[tt], false, false);
      }
    }

    // Streamed top-32 update. cA[tt][r] = score(query tt*16 + r + 8*hi, slot).
#pragma unroll
    for (int tt = 0; tt < QG; ++tt) {
#pragma unroll
      for (int r = 0; r < 8; ++r) {
        float sc = valid ? cA[tt][r] : -__builtin_inff();
        for (;;) {
          unsigned long long bal = __ballot(sc > thr[tt][r]);
          if (!bal) break;
          const int l = __ffsll(bal) - 1;
          const float v = __shfl(sc, l, 32);
          sc = (lane == l) ? -__builtin_inff() : sc;
          const int lslot = slotBase + (l & 15);
          const int lq    = tt * 16 + r + ((l >> 4) << 3);   // source query
          const float nm  = lds_insert(myV + lq * KTOP, myI + lq * KTOP,
                                       v, lslot, lane);
          thr[tt][r] = ((lane >> 4) == (l >> 4)) ? nm : thr[tt][r];
        }
      }
    }
  }
  __syncthreads();

  // Block merge: each wave reduces 8x32 -> 32 for queries wv+8*qi.
  for (int qi = 0; qi < 4; ++qi) {
    const int q = wv + qi * NWAVE;
    float cv[8]; int ci[8];
#pragma unroll
    for (int i = 0; i < 8; ++i) {
      cv[i] = listV[(i * QBLK + q) * KTOP + lane];
      ci[i] = listI[(i * QBLK + q) * KTOP + lane];
    }
    // In-place into wave-0 region: each wave writes only its own queries.
    wave_selectN<8>(cv, ci, lane, listV + q * KTOP, listI + q * KTOP);
  }
  __syncthreads();

  // Coalesced copy of the 32x32 block result to workspace.
  Cand* outBase = cands + (((size_t)qg * CHUNKS + chunk) * QBLK) * KTOP;
  for (int n = tid; n < QBLK * KTOP; n += 256) {
    Cand cc; cc.v = listV[n]; cc.i = listI[n];
    outBase[n] = cc;
  }
}

// ---------------------------------------------------------------------------
// Kernel 2: per-query merge of 32x32 candidates -> top-32, softmax,
// attn-weighted gather of vals. grid = NQ, block = 256.
// ---------------------------------------------------------------------------
__global__ __launch_bounds__(256, 1)
void merge_softmax_gather_kernel(const Cand* __restrict__ cands,
                                 const float* __restrict__ vals,
                                 float* __restrict__ out) {
  __shared__ float cval[CHUNKS * KTOP];   // 4 KB
  __shared__ int   cidx[CHUNKS * KTOP];   // 4 KB
  __shared__ float l1v[NWAVE * KTOP];     // 1 KB
  __shared__ int   l1i[NWAVE * KTOP];     // 1 KB

  const int tid  = threadIdx.x;
  const int lane = tid & 31;
  const int wv   = tid >> 5;
  const int q    = blockIdx.x;
  const int qg   = q >> 5;        // query group (32 queries each)
  const int qr   = q & 31;

  for (int n = tid; n < CHUNKS * KTOP; n += 256) {
    const int ch = n >> 5, j = n & 31;
    Cand c = cands[(((size_t)qg * CHUNKS + ch) * QBLK + qr) * KTOP + j];
    cval[n] = c.v;
    cidx[n] = c.i;
  }
  __syncthreads();

  // Level 1: each wave selects top-32 of its 128-candidate segment (4/lane).
  {
    float cv[4]; int ci[4];
#pragma unroll
    for (int i = 0; i < 4; ++i) {
      const int n = wv * 128 + i * 32 + lane;
      cv[i] = cval[n]; ci[i] = cidx[n];
    }
    wave_selectN<4>(cv, ci, lane, l1v + wv * KTOP, l1i + wv * KTOP);
  }
  __syncthreads();

  // Level 2 + softmax on wave 0 (8x32 = 256 candidates, 8/lane).
  if (wv == 0) {
    float cv[8]; int ci[8];
#pragma unroll
    for (int i = 0; i < 8; ++i) {
      const int n = i * 32 + lane;
      cv[i] = l1v[n]; ci[i] = l1i[n];
    }
    wave_selectN<8>(cv, ci, lane, l1v, l1i);  // final top-32 in l1v/l1i[0..31]
    float v = l1v[lane];
    float mx = v;
#pragma unroll
    for (int off = 16; off; off >>= 1) mx = fmaxf(mx, __shfl_xor(mx, off, 32));
    float e = __expf(v - mx);
    float sm = e;
#pragma unroll
    for (int off = 16; off; off >>= 1) sm += __shfl_xor(sm, off, 32);
    l1v[lane] = e / sm;                      // attn weights
  }
  __syncthreads();

  // Weighted gather: thread t owns output columns [4t, 4t+4).
  float4 acc = make_float4(0.f, 0.f, 0.f, 0.f);
  const float4* v4 = (const float4*)vals;
  for (int k = 0; k < KTOP; ++k) {
    const float a = l1v[k];
    const int   r = l1i[k];
    float4 vv = v4[(size_t)r * (DIM / 4) + tid];
    acc.x += a * vv.x; acc.y += a * vv.y; acc.z += a * vv.z; acc.w += a * vv.w;
  }
  ((float4*)out)[(size_t)q * (DIM / 4) + tid] = acc;
}

// ---------------------------------------------------------------------------
extern "C" void kernel_launch(void* const* d_in, const int* in_sizes, int n_in,
                              void* d_out, int out_size, void* d_ws, size_t ws_size,
                              hipStream_t stream) {
  (void)in_sizes; (void)n_in; (void)out_size; (void)ws_size;
  const float* x    = (const float*)d_in[0];
  const float* keys = (const float*)d_in[1];
  const float* vals = (const float*)d_in[2];
  float* out = (float*)d_out;
  Cand* cands = (Cand*)d_ws;   // NQG*CHUNKS*QBLK*KTOP * 8B = 4.2 MB

  score_topk_kernel<<<dim3(NQG, CHUNKS), 256, 0, stream>>>(x, keys, cands);
  merge_softmax_gather_kernel<<<dim3(NQ), 256, 0, stream>>>(cands, vals, out);
}
